// ConditionalGraphAttentionGeneration_9655086481444
// MI455X (gfx1250) — matrix operand order repaired and checked by reading, not compiled, not run on hardware
//
#include <hip/hip_runtime.h>

#define B_   32
#define N_   8192
#define QD_  256
#define HD_  256
#define NH_  4
#define DH_  64
#define RT_  4            // row-tiles (of 128 rows) per block

typedef __attribute__((ext_vector_type(16))) _Float16 v16h;
typedef __attribute__((ext_vector_type(8)))  float    v8f;

// Dynamic LDS (132-135 KB; static __shared__ would trip the compiler limit).
extern __shared__ char cga_smem[];

// ---------------------------------------------------------------------------
// A fragment for v_wmma_f32_16x16x32_f16 (wave32), ISA layout:
// lane holds row m = lane&15; lanes 0-15: K 0..7 & 16..23, lanes 16-31: K 8..15 & 24..31.
// Source row-major fp32, converted to f16 in-register.
__device__ inline v16h load_a_f32(const float* __restrict__ p0, int ld, int lane) {
  const int m  = lane & 15;
  const int kb = (lane & 16) ? 8 : 0;
  const float* p = p0 + (size_t)m * ld + kb;
  v16h a;
#pragma unroll
  for (int i = 0; i < 8; ++i) a[i]     = (_Float16)p[i];
#pragma unroll
  for (int i = 0; i < 8; ++i) a[i + 8] = (_Float16)p[16 + i];
  return a;
}

// B fragment from LDS, fragment-major pre-swizzled layout: fragment f=kc*16+t,
// each lane's 16 halfs contiguous at ((f*32+lane)*16).
__device__ inline v16h load_b_lds(const _Float16* lw, int kc, int t, int lane) {
  const _Float16* p = lw + (((kc * 16 + t) * 32 + lane) << 4);
  v16h b;
#pragma unroll
  for (int i = 0; i < 16; ++i) b[i] = p[i];
  return b;
}

// ---------------------------------------------------------------------------
// Stage a byte blob (the pre-swizzled weights) into LDS via the Tensor Data
// Mover (async DMA, TENSORcnt) when available; cooperative copy otherwise.
#if __has_builtin(__builtin_amdgcn_tensor_load_to_lds)
typedef __attribute__((ext_vector_type(4))) unsigned int u32x4;
typedef __attribute__((ext_vector_type(8))) int          i32x8;
typedef __attribute__((ext_vector_type(4))) int          i32x4;

__device__ inline void stage_to_lds(const void* src, void* ldsdst, unsigned bytes) {
  if (threadIdx.x < 32) {                 // one wave issues the TDM op
    const unsigned long long ga = (unsigned long long)(__SIZE_TYPE__)src;
    const unsigned lds = (unsigned)(__SIZE_TYPE__)ldsdst;
    const unsigned n4  = bytes >> 2;      // 4-byte elements, flat 1-row tile
    u32x4 g0;
    g0[0] = 1u;                                               // count=1
    g0[1] = lds;                                              // lds_addr
    g0[2] = (unsigned)(ga & 0xffffffffu);                     // global_addr lo
    g0[3] = (unsigned)((ga >> 32) & 0x1ffffffu) | (2u << 30); // addr[56:32]|type=2
    i32x8 g1;
    g1[0] = (int)(2u << 16);                                  // data_size=4B
    g1[1] = (int)((n4 & 0xffffu) << 16);                      // tensor_dim0 lo
    g1[2] = (int)(((n4 >> 16) & 0xffffu) | (1u << 16));       // dim0 hi | dim1=1
    g1[3] = (int)((n4 & 0xffffu) << 16);                      // tile_dim0=n4
    g1[4] = 1;                                                // tile_dim1=1
    g1[5] = (int)n4;                                          // dim0_stride lo
    g1[6] = 0;
    g1[7] = 0;
    i32x4 z4 = {0, 0, 0, 0};
    i32x8 z8 = {0, 0, 0, 0, 0, 0, 0, 0};
    __builtin_amdgcn_tensor_load_to_lds(g0, g1, z4, z4, z8, 0);
  }
#if __has_builtin(__builtin_amdgcn_s_wait_tensorcnt)
  __builtin_amdgcn_s_wait_tensorcnt(0);
#endif
}
#else
__device__ inline void stage_to_lds(const void* src, void* ldsdst, unsigned bytes) {
  const float4* s = (const float4*)src;
  float4* d = (float4*)ldsdst;
  const int n = (int)(bytes >> 4);
  for (int i = threadIdx.x; i < n; i += 256) d[i] = s[i];
}
#endif

// ---------------------------------------------------------------------------
// Kernel 0: convert 256x256 fp32 weight to f16 *fragment-major* blob:
// fragment f = kc*16 + t ; lane-slot (f*32+lane) holds 16 contiguous halfs:
// element i -> W[(kc*32 + 8*(lane>=16) + (i&7) + 16*(i>=8))*256 + t*16 + (lane&15)].
__global__ __launch_bounds__(256)
void cga_convert_frag(const float* __restrict__ W, _Float16* __restrict__ WF) {
  const int id   = blockIdx.x * 256 + threadIdx.x;  // 0..4095 lane-slots
  const int lane = id & 31;
  const int f    = id >> 5;
  const int kc   = f >> 4;
  const int t    = f & 15;
  const int k0   = kc * 32 + ((lane & 16) ? 8 : 0);
  const int col  = t * 16 + (lane & 15);
  _Float16* out = WF + (size_t)id * 16;
#pragma unroll
  for (int i = 0; i < 16; ++i)
    out[i] = (_Float16)W[(size_t)(k0 + (i & 7) + ((i & 8) ? 16 : 0)) * HD_ + col];
}

// Kernel 1: hq = query@Wq + bq (tiny), also zero scored accumulator.
__global__ __launch_bounds__(256)
void cga_hq(const float* __restrict__ q, const float* __restrict__ Wq,
            const float* __restrict__ bq, float* __restrict__ hq,
            float* __restrict__ scored) {
  const int b = blockIdx.x, j = threadIdx.x;
  __shared__ float qrow[QD_];
  qrow[j] = q[(size_t)b * QD_ + j];
  __syncthreads();
  float acc = bq[j];
#pragma unroll 4
  for (int k = 0; k < QD_; ++k) acc = fmaf(qrow[k], Wq[(size_t)k * HD_ + j], acc);
  hq[(size_t)b * HD_ + j] = acc;
  scored[(size_t)b * HD_ + j] = 0.f;
}

// ---------------------------------------------------------------------------
// Software-pipelined 16x256 = A(16x256) x W(256x256) tile GEMM.
// A double-buffered across K-chunks (hides HBM latency), B double-buffered
// across n-tiles (hides LDS latency behind the matrix pipe).
__device__ inline void tile_gemm(const float* __restrict__ arow,
                                 const _Float16* lds_w, int lane, v8f acc[16]) {
  v16h af = load_a_f32(arow, QD_, lane);
#pragma unroll 1
  for (int kk = 0; kk < QD_; kk += 32) {
    const v16h cur_a = af;
    if (kk + 32 < QD_) af = load_a_f32(arow + kk + 32, QD_, lane);
    const int kc = kk >> 5;
    v16h bf = load_b_lds(lds_w, kc, 0, lane);
#pragma unroll
    for (int t = 0; t < 16; ++t) {
      const v16h cur_b = bf;
      if (t < 15) bf = load_b_lds(lds_w, kc, t + 1, lane);
      acc[t] = __builtin_amdgcn_wmma_f32_16x16x32_f16(
          false, cur_a, false, cur_b, (short)0, acc[t], false, false);
    }
  }
}

// ---------------------------------------------------------------------------
// Kernel 2: hk = key@Wq via WMMA (weights LDS-resident), fused
// leaky_relu + a-dot score, raw scores into the e output region.
__global__ __launch_bounds__(256)
void cga_score(const float* __restrict__ key, const _Float16* __restrict__ WqF,
               const float* __restrict__ hq, const float* __restrict__ bq,
               const float* __restrict__ avec, float* __restrict__ e_out) {
  const int b    = blockIdx.y;
  const int wave = threadIdx.x >> 5;
  const int lane = threadIdx.x & 31;

  _Float16* lds_w = (_Float16*)cga_smem;                 // 131072 B
  float*    c_lds = (float*)(cga_smem + 131072);         // 1024 B
  float*    a_lds = (float*)(cga_smem + 131072 + 1024);  // 256 B

  stage_to_lds(WqF, lds_w, QD_ * HD_ * 2);
  c_lds[threadIdx.x] = hq[(size_t)b * HD_ + threadIdx.x] + bq[threadIdx.x];
  if (threadIdx.x < DH_) a_lds[threadIdx.x] = avec[threadIdx.x];
  __syncthreads();

  const int nloc    = lane & 15;
  const int halfsel = (lane >> 4) & 1;

  for (int it = 0; it < RT_; ++it) {
    const int row0 = (blockIdx.x * RT_ + it) * 128 + wave * 16;
    const float* keyb = key + ((size_t)b * N_ + row0) * QD_;

    // prefetch next row-tile's A fragment region (stays in-bounds)
    if (blockIdx.x * RT_ + it + 1 < N_ / 128)
      __builtin_prefetch(keyb + (size_t)128 * QD_ + (size_t)(lane & 15) * QD_, 0, 0);

    v8f acc[16];
    v8f zero = {};
#pragma unroll
    for (int t = 0; t < 16; ++t) acc[t] = zero;

    tile_gemm(keyb, lds_w, lane, acc);

    // score: s[row,h] = sum_col lrelu(hq+bq+hk)*a ; cross-lane reduce.
#pragma unroll
    for (int h = 0; h < NH_; ++h) {
      float sr[8];
#pragma unroll
      for (int r = 0; r < 8; ++r) sr[r] = 0.f;
#pragma unroll
      for (int tt = 0; tt < 4; ++tt) {
        const int t   = h * 4 + tt;
        const int col = t * 16 + nloc;
        const float cv = c_lds[col];
        const float av = a_lds[col & (DH_ - 1)];
#pragma unroll
        for (int r = 0; r < 8; ++r) {
          float v = acc[t][r] + cv;
          v = (v > 0.f) ? v : 0.2f * v;
          sr[r] = fmaf(v, av, sr[r]);
        }
      }
#pragma unroll
      for (int r = 0; r < 8; ++r) {
        float v = sr[r];
        v += __shfl_xor(v, 1, 32);
        v += __shfl_xor(v, 2, 32);
        v += __shfl_xor(v, 4, 32);
        v += __shfl_xor(v, 8, 32);
        if (nloc == 0)
          e_out[((size_t)b * N_ + row0 + halfsel * 8 + r) * NH_ + h] = v;
      }
    }
  }
}

// ---------------------------------------------------------------------------
// Kernel 3: softmax statistics per (b,h).
__global__ __launch_bounds__(256)
void cga_stats(const float* __restrict__ s, float* __restrict__ mbuf,
               float* __restrict__ lbuf) {
  const int b = blockIdx.x >> 2;
  const int h = blockIdx.x & 3;
  const float* sp = s + (size_t)b * N_ * NH_ + h;
  __shared__ float red[256];
  float mx = -3.4e38f;
  for (int n = threadIdx.x; n < N_; n += 256) mx = fmaxf(mx, sp[(size_t)n * NH_]);
  red[threadIdx.x] = mx;
  __syncthreads();
  for (int off = 128; off > 0; off >>= 1) {
    if (threadIdx.x < off) red[threadIdx.x] = fmaxf(red[threadIdx.x], red[threadIdx.x + off]);
    __syncthreads();
  }
  mx = red[0];
  __syncthreads();
  float sum = 0.f;
  for (int n = threadIdx.x; n < N_; n += 256) sum += __expf(sp[(size_t)n * NH_] - mx);
  red[threadIdx.x] = sum;
  __syncthreads();
  for (int off = 128; off > 0; off >>= 1) {
    if (threadIdx.x < off) red[threadIdx.x] += red[threadIdx.x + off];
    __syncthreads();
  }
  if (threadIdx.x == 0) { mbuf[blockIdx.x] = mx; lbuf[blockIdx.x] = red[0]; }
}

// ---------------------------------------------------------------------------
// Kernel 4: hv = value@Wv via WMMA, normalize e in place, weighted row
// reduction accumulated in LDS across all row-tiles, then one global atomic.
__global__ __launch_bounds__(256)
void cga_value(const float* __restrict__ value, const _Float16* __restrict__ WvF,
               const float* __restrict__ bv, const float* __restrict__ mbuf,
               const float* __restrict__ lbuf, float* __restrict__ e_io,
               float* __restrict__ scored) {
  const int b    = blockIdx.y;
  const int wave = threadIdx.x >> 5;
  const int lane = threadIdx.x & 31;

  _Float16* lds_w  = (_Float16*)cga_smem;                        // 131072 B
  float*    bv_lds = (float*)(cga_smem + 131072);                // 1024 B
  float*    red    = (float*)(cga_smem + 131072 + 1024);         // 1024 B
  float*    w_lds  = (float*)(cga_smem + 131072 + 2048);         // 8*16*4 floats

  stage_to_lds(WvF, lds_w, QD_ * HD_ * 2);
  bv_lds[threadIdx.x] = bv[threadIdx.x];
  red[threadIdx.x] = 0.f;
  __syncthreads();

  const int nloc    = lane & 15;
  const int halfsel = (lane >> 4) & 1;

  for (int it = 0; it < RT_; ++it) {
    const int row0 = (blockIdx.x * RT_ + it) * 128 + wave * 16;
    const float* valb = value + ((size_t)b * N_ + row0) * QD_;

    if (blockIdx.x * RT_ + it + 1 < N_ / 128)
      __builtin_prefetch(valb + (size_t)128 * QD_ + (size_t)(lane & 15) * QD_, 0, 0);

    // softmax weights for this wave's 16 rows x 4 heads (same-wave LDS use).
#pragma unroll
    for (int i = 0; i < 2; ++i) {
      const int idx = lane * 2 + i;          // 0..63
      const int m = idx >> 2, h = idx & 3;
      const size_t off = ((size_t)b * N_ + row0 + m) * NH_ + h;
      const float w = __expf(e_io[off] - mbuf[b * NH_ + h]) / lbuf[b * NH_ + h];
      e_io[off] = w;
      w_lds[(wave * 16 + m) * NH_ + h] = w;
    }

    v8f acc[16];
    v8f zero = {};
#pragma unroll
    for (int t = 0; t < 16; ++t) acc[t] = zero;

    tile_gemm(valb, lds_w, lane, acc);

#pragma unroll
    for (int t = 0; t < 16; ++t) {
      const int col = t * 16 + nloc;
      const int h   = t >> 2;
      const float bvv = bv_lds[col];
      float partial = 0.f;
#pragma unroll
      for (int r = 0; r < 8; ++r)
        partial = fmaf(w_lds[(wave * 16 + halfsel * 8 + r) * NH_ + h],
                       acc[t][r] + bvv, partial);
      partial += __shfl_xor(partial, 16, 32);       // fold rows 8-15 into 0-7
      if (lane < 16) atomicAdd(&red[col], partial); // ds_add_f32, accum over it
    }
  }
  __syncthreads();
  atomicAdd(&scored[(size_t)b * HD_ + threadIdx.x], red[threadIdx.x]);
}

// Kernel 5: h_prime = relu(scored)
__global__ __launch_bounds__(256)
void cga_finalize(const float* __restrict__ scored, float* __restrict__ out) {
  const int i = blockIdx.x * 256 + threadIdx.x;
  out[i] = fmaxf(scored[i], 0.f);
}

// ---------------------------------------------------------------------------
extern "C" void kernel_launch(void* const* d_in, const int* in_sizes, int n_in,
                              void* d_out, int out_size, void* d_ws, size_t ws_size,
                              hipStream_t stream) {
  const float* query = (const float*)d_in[0];
  const float* key   = (const float*)d_in[1];
  const float* value = (const float*)d_in[2];
  const float* Wq    = (const float*)d_in[3];
  const float* bq    = (const float*)d_in[4];
  const float* Wv    = (const float*)d_in[5];
  const float* bv    = (const float*)d_in[6];
  const float* avec  = (const float*)d_in[7];

  char* ws = (char*)d_ws;
  _Float16* WqF    = (_Float16*)(ws);              // 131072 B, fragment-major
  _Float16* WvF    = (_Float16*)(ws + 131072);     // 131072 B
  float*    hq     = (float*)(ws + 262144);        // 32768 B
  float*    scored = (float*)(ws + 294912);        // 32768 B
  float*    mbuf   = (float*)(ws + 327680);        // 512 B
  float*    lbuf   = (float*)(ws + 328192);        // 512 B

  float* hprime = (float*)d_out;                   // [B, H*D] = 8192 floats
  float* e_out  = hprime + (size_t)B_ * HD_;       // [B, N, H] = 1048576 floats

  cga_convert_frag<<<16, 256, 0, stream>>>(Wq, WqF);
  cga_convert_frag<<<16, 256, 0, stream>>>(Wv, WvF);
  cga_hq<<<B_, HD_, 0, stream>>>(query, Wq, bq, hq, scored);

  dim3 gtile(N_ / (128 * RT_), B_);
  const size_t smem_score = 131072 + 1024 + 256;
  const size_t smem_value = 131072 + 1024 + 1024 + 8 * 16 * NH_ * 4;
  cga_score<<<gtile, 256, smem_score, stream>>>(key, WqF, hq, bq, avec, e_out);
  cga_stats<<<B_ * NH_, 256, 0, stream>>>(e_out, mbuf, lbuf);
  cga_value<<<gtile, 256, smem_value, stream>>>(value, WvF, bv, mbuf, lbuf, e_out, scored);
  cga_finalize<<<B_, HD_, 0, stream>>>(scored, hprime);
}
